// HQQLinear_44349832298868
// MI455X (gfx1250) — matrix operand order repaired and checked by reading, not compiled, not run on hardware
//
#include <hip/hip_runtime.h>

// ---- vector types ----
typedef __attribute__((ext_vector_type(16))) _Float16 v16h;
typedef __attribute__((ext_vector_type(8)))  _Float16 v8h;
typedef __attribute__((ext_vector_type(4)))  _Float16 v4h;
typedef __attribute__((ext_vector_type(2)))  __fp16   v2fp;   // return type of cvt_pkrtz
typedef __attribute__((ext_vector_type(8)))  float    v8f;
typedef __attribute__((ext_vector_type(4)))  float    f32x4;
typedef __attribute__((ext_vector_type(4)))  int      i32x4;
typedef __attribute__((ext_vector_type(4)))  unsigned int u32x4;
typedef __attribute__((ext_vector_type(8)))  int      i32x8v;
typedef __attribute__((ext_vector_type(4)))  int      i32x4v;

#define BM 256
#define BN 128
#define BK 64          // == GROUP_SIZE
#define KDIM 4096
#define NDIM 4096
#define MDIM 8192
#define NT   (KDIM / BK)
#define LDSS 72        // halfs per LDS row: 64 + 8 pad -> 144 B (16B-aligned stride)

union V16U { v16h v; v8h h[2]; };
union V4U  { v4h  v; v2fp h[2]; };

// ============================================================================
// Pass 1a: x f32 -> f16  (8 elems / thread)
// ============================================================================
__global__ __launch_bounds__(256)
void cvt_x_kernel(const float* __restrict__ x, _Float16* __restrict__ xh)
{
    const size_t i = ((size_t)blockIdx.x * 256 + threadIdx.x) * 8;
    const f32x4 a = *(const f32x4*)(x + i);
    const f32x4 b = *(const f32x4*)(x + i + 4);
    V4U u0, u1;
    u0.h[0] = __builtin_amdgcn_cvt_pkrtz(a.x, a.y);
    u0.h[1] = __builtin_amdgcn_cvt_pkrtz(a.z, a.w);
    u1.h[0] = __builtin_amdgcn_cvt_pkrtz(b.x, b.y);
    u1.h[1] = __builtin_amdgcn_cvt_pkrtz(b.z, b.w);
    *(v4h*)(xh + i)     = u0.v;
    *(v4h*)(xh + i + 4) = u1.v;
}

// ============================================================================
// Pass 1b: W nested-dequant int32(4-bit vals) -> f16  (16 elems / thread)
// ============================================================================
__global__ __launch_bounds__(256)
void deq_w_kernel(const int* __restrict__ Wq,
                  const int* __restrict__ scale_q,
                  const int* __restrict__ zero_q,
                  const float* __restrict__ s_scale_p,
                  const float* __restrict__ z_scale_p,
                  const float* __restrict__ s_zero_p,
                  const float* __restrict__ z_zero_p,
                  _Float16* __restrict__ wh)
{
    const int t = blockIdx.x * 256 + threadIdx.x;   // 0..1048575
    const int g = t >> 2;                           // group id
    const int c = (t & 3) * 16;                     // element offset in group

    const float s_scale = *s_scale_p;
    const float s_zero  = *s_zero_p;
    const float c_s     = -(*z_scale_p) * s_scale;
    const float c_z     = -(*z_zero_p)  * s_zero;

    const float scale = __builtin_fmaf((float)scale_q[g], s_scale, c_s);
    const float zero  = __builtin_fmaf((float)zero_q[g],  s_zero,  c_z);
    const float mzs   = -zero * scale;

    const int* src = Wq + (size_t)g * 64 + c;
    _Float16*  dst = wh + (size_t)g * 64 + c;
    #pragma unroll
    for (int j = 0; j < 4; ++j) {
        const i32x4 qv = *(const i32x4*)(src + j * 4);
        V4U u;
        u.h[0] = __builtin_amdgcn_cvt_pkrtz(
                     __builtin_fmaf((float)qv.x, scale, mzs),
                     __builtin_fmaf((float)qv.y, scale, mzs));
        u.h[1] = __builtin_amdgcn_cvt_pkrtz(
                     __builtin_fmaf((float)qv.z, scale, mzs),
                     __builtin_fmaf((float)qv.w, scale, mzs));
        *(v4h*)(dst + j * 4) = u.v;
    }
}

// ============================================================================
// Pass 2: f16 GEMM, double-buffered LDS tiles fed by the Tensor Data Mover
// ============================================================================
// D# group0: count=1 | lds_addr | global_addr(57b) | type=2
static __device__ __forceinline__ u32x4 make_g0(unsigned lds_addr,
                                                unsigned long long gaddr)
{
    u32x4 g0;
    g0.x = 1u;                                        // count=1, user mode
    g0.y = lds_addr;                                  // LDS byte address
    g0.z = (unsigned)gaddr;                           // global_addr[31:0]
    g0.w = ((unsigned)(gaddr >> 32) & 0x01FFFFFFu)    // global_addr[56:32]
           | (2u << 30);                              // type=2 (image)
    return g0;
}

// D# group1: 2-D tile (tile_dim0 x tile_dim1) of 2-byte elems with LDS padding
// pad: every 128 B (interval code 4) insert 16 B (amount code 3) -> 144 B rows
static __device__ __forceinline__ i32x8v make_g1(int tensor_d0, int tensor_d1,
                                                 int tile_d0, int tile_d1,
                                                 int d0_stride)
{
    i32x8v g1;
    g1[0] = (1 << 16) | (1 << 20) | (4 << 22) | (3 << 25); // data_size=2B, pad on
    g1[1] = (tensor_d0 & 0xffff) << 16;                    // tensor_dim0[15:0]
    g1[2] = ((tensor_d0 >> 16) & 0xffff) | ((tensor_d1 & 0xffff) << 16);
    g1[3] = ((tensor_d1 >> 16) & 0xffff) | ((tile_d0 & 0xffff) << 16);
    g1[4] = (tile_d1 & 0xffff);                            // tile_dim1, tile_dim2=0
    g1[5] = d0_stride;                                     // dim0 stride low 32
    g1[6] = 0;                                             // stride hi / dim1 stride
    g1[7] = 0;
    return g1;
}

__global__ __launch_bounds__(256)
void gemm_tdm_kernel(const _Float16* __restrict__ xh,
                     const _Float16* __restrict__ wh,
                     const float* __restrict__ bias,
                     float* __restrict__ out)
{
    __shared__ _Float16 As[2][BM * LDSS];   // double-buffered, TDM-filled
    __shared__ _Float16 Bs[2][BN * LDSS];

    const int tid  = threadIdx.x;
    const int lane = tid & 31;
    const int hh   = lane >> 4;
    const int r    = lane & 15;
    const int wave = tid >> 5;
    const int wm   = (wave >> 1) * 64;
    const int wn   = (wave & 1) * 64;

    const int m0 = blockIdx.y * BM;
    const int n0 = blockIdx.x * BN;

    const unsigned asAddr0 = (unsigned)(size_t)(&As[0][0]);
    const unsigned asAddr1 = (unsigned)(size_t)(&As[1][0]);
    const unsigned bsAddr0 = (unsigned)(size_t)(&Bs[0][0]);
    const unsigned bsAddr1 = (unsigned)(size_t)(&Bs[1][0]);

    unsigned long long gax = (unsigned long long)(size_t)(xh + (size_t)m0 * KDIM);
    unsigned long long gaw = (unsigned long long)(size_t)(wh + (size_t)n0 * KDIM);

    const i32x8v g1x = make_g1(KDIM, MDIM, BK, BM, KDIM);
    const i32x8v g1w = make_g1(KDIM, NDIM, BK, BN, KDIM);
    const i32x4v gz4 = {0, 0, 0, 0};
    const i32x8v gz8 = {0, 0, 0, 0, 0, 0, 0, 0};

    v8f acc[4][4];
    #pragma unroll
    for (int i = 0; i < 4; ++i)
        #pragma unroll
        for (int j = 0; j < 4; ++j)
            #pragma unroll
            for (int e = 0; e < 8; ++e)
                acc[i][j][e] = 0.0f;

    // prologue: tile 0 -> buffer 0
    if (wave == 0) {
        __builtin_amdgcn_tensor_load_to_lds(make_g0(asAddr0, gax), g1x,
                                            gz4, gz4, gz8, 0);
        __builtin_amdgcn_tensor_load_to_lds(make_g0(bsAddr0, gaw), g1w,
                                            gz4, gz4, gz8, 0);
    }

    for (int kt = 0; kt < NT; ++kt) {
        if (wave == 0) {
            if (kt + 1 < NT) {
                // prefetch tile kt+1 into the other buffer, then wait for tile kt
                // (TDM ops complete in order -> cnt<=2 means first pair landed)
                if ((kt + 1) & 1) {
                    __builtin_amdgcn_tensor_load_to_lds(make_g0(asAddr1, gax + 128),
                                                        g1x, gz4, gz4, gz8, 0);
                    __builtin_amdgcn_tensor_load_to_lds(make_g0(bsAddr1, gaw + 128),
                                                        g1w, gz4, gz4, gz8, 0);
                } else {
                    __builtin_amdgcn_tensor_load_to_lds(make_g0(asAddr0, gax + 128),
                                                        g1x, gz4, gz4, gz8, 0);
                    __builtin_amdgcn_tensor_load_to_lds(make_g0(bsAddr0, gaw + 128),
                                                        g1w, gz4, gz4, gz8, 0);
                }
                __builtin_amdgcn_s_wait_tensorcnt(2);
            } else {
                __builtin_amdgcn_s_wait_tensorcnt(0);
            }
        }
        __syncthreads();            // tile kt visible to all waves

        const _Float16* Ab = As[kt & 1];
        const _Float16* Bb = Bs[kt & 1];

        #pragma unroll
        for (int ks = 0; ks < 2; ++ks) {
            const int kk = ks * 32;

            v16h af[4];
            #pragma unroll
            for (int i = 0; i < 4; ++i) {
                const _Float16* src = Ab + (wm + i * 16 + r) * LDSS + kk + hh * 8;
                V16U u;
                u.h[0] = *(const v8h*)(src);
                u.h[1] = *(const v8h*)(src + 16);
                af[i] = u.v;
            }
            v16h bfr[4];
            #pragma unroll
            for (int j = 0; j < 4; ++j) {
                const _Float16* src = Bb + (wn + j * 16 + r) * LDSS + kk + hh * 8;
                V16U u;
                u.h[0] = *(const v8h*)(src);
                u.h[1] = *(const v8h*)(src + 16);
                bfr[j] = u.v;
            }
            #pragma unroll
            for (int i = 0; i < 4; ++i)
                #pragma unroll
                for (int j = 0; j < 4; ++j)
                    acc[i][j] = __builtin_amdgcn_wmma_f32_16x16x32_f16(
                        false, af[i], false, bfr[j],
                        (short)0, acc[i][j], false, false);
        }

        __syncthreads();            // reads done before this buffer is re-filled
        gax += BK * 2;              // advance 64 f16 = 128 bytes
        gaw += BK * 2;
    }

    float bv[4];
    #pragma unroll
    for (int j = 0; j < 4; ++j)
        bv[j] = bias[n0 + wn + j * 16 + r];

    #pragma unroll
    for (int i = 0; i < 4; ++i)
        #pragma unroll
        for (int j = 0; j < 4; ++j) {
            float* o = out + (size_t)(m0 + wm + i * 16 + hh * 8) * NDIM
                           + (n0 + wn + j * 16 + r);
            #pragma unroll
            for (int v = 0; v < 8; ++v)
                __builtin_nontemporal_store(acc[i][j][v] + bv[j], o + (size_t)v * NDIM);
        }
}

// ============================================================================
// Fallback: fused dequant GEMM (used when workspace is too small)
// ============================================================================
__global__ __launch_bounds__(256)
void hqq_wmma_fused_kernel(const float* __restrict__ x,
                           const int*   __restrict__ Wq,
                           const int*   __restrict__ scale_q,
                           const int*   __restrict__ zero_q,
                           const float* __restrict__ s_scale_p,
                           const float* __restrict__ z_scale_p,
                           const float* __restrict__ s_zero_p,
                           const float* __restrict__ z_zero_p,
                           const float* __restrict__ bias,
                           float* __restrict__ out)
{
    __shared__ _Float16 As[BM * LDSS];
    __shared__ _Float16 Bs[BN * LDSS];

    const int tid  = threadIdx.x;
    const int lane = tid & 31;
    const int hh   = lane >> 4;
    const int r    = lane & 15;
    const int wave = tid >> 5;
    const int wm   = (wave >> 1) * 64;
    const int wn   = (wave & 1) * 64;
    const int m0 = blockIdx.y * BM;
    const int n0 = blockIdx.x * BN;

    const float s_scale = *s_scale_p;
    const float s_zero  = *s_zero_p;
    const float c_s     = -(*z_scale_p) * s_scale;
    const float c_z     = -(*z_zero_p)  * s_zero;

    const int rx = tid >> 4;
    const int cx = tid & 15;
    const float* xp = x + (size_t)(m0 + rx) * KDIM + cx * 4;

    const int rw  = tid >> 1;
    const int hw8 = (tid & 1) * 8;
    const int*  wp = Wq      + (size_t)(n0 + rw) * 64 * 64 + hw8 * 4;
    const int*  sp = scale_q + (size_t)(n0 + rw) * 64;
    const int*  zp = zero_q  + (size_t)(n0 + rw) * 64;

    v8f acc[4][4];
    #pragma unroll
    for (int i = 0; i < 4; ++i)
        #pragma unroll
        for (int j = 0; j < 4; ++j)
            #pragma unroll
            for (int e = 0; e < 8; ++e)
                acc[i][j][e] = 0.0f;

    for (int kt = 0; kt < NT; ++kt) {
        const float scale = __builtin_fmaf((float)(*sp), s_scale, c_s);
        const float zero  = __builtin_fmaf((float)(*zp), s_zero,  c_z);
        const float mzs   = -zero * scale;

        #pragma unroll
        for (int j = 0; j < 16; ++j) {
            const f32x4 xv = *(const f32x4*)(xp + (size_t)j * 16 * KDIM);
            V4U u;
            u.h[0] = __builtin_amdgcn_cvt_pkrtz(xv.x, xv.y);
            u.h[1] = __builtin_amdgcn_cvt_pkrtz(xv.z, xv.w);
            *(v4h*)(As + (rx + 16 * j) * LDSS + cx * 4) = u.v;
        }
        #pragma unroll
        for (int j = 0; j < 8; ++j) {
            const i32x4 qv = *(const i32x4*)(wp + j * 4);
            V4U u;
            u.h[0] = __builtin_amdgcn_cvt_pkrtz(
                         __builtin_fmaf((float)qv.x, scale, mzs),
                         __builtin_fmaf((float)qv.y, scale, mzs));
            u.h[1] = __builtin_amdgcn_cvt_pkrtz(
                         __builtin_fmaf((float)qv.z, scale, mzs),
                         __builtin_fmaf((float)qv.w, scale, mzs));
            *(v4h*)(Bs + rw * LDSS + (hw8 + j) * 4) = u.v;
        }
        __syncthreads();

        #pragma unroll
        for (int ks = 0; ks < 2; ++ks) {
            const int kk = ks * 32;
            v16h af[4];
            #pragma unroll
            for (int i = 0; i < 4; ++i) {
                const _Float16* src = As + (wm + i * 16 + r) * LDSS + kk + hh * 8;
                V16U u;
                u.h[0] = *(const v8h*)(src);
                u.h[1] = *(const v8h*)(src + 16);
                af[i] = u.v;
            }
            v16h bfr[4];
            #pragma unroll
            for (int j = 0; j < 4; ++j) {
                const _Float16* src = Bs + (wn + j * 16 + r) * LDSS + kk + hh * 8;
                V16U u;
                u.h[0] = *(const v8h*)(src);
                u.h[1] = *(const v8h*)(src + 16);
                bfr[j] = u.v;
            }
            #pragma unroll
            for (int i = 0; i < 4; ++i)
                #pragma unroll
                for (int j = 0; j < 4; ++j)
                    acc[i][j] = __builtin_amdgcn_wmma_f32_16x16x32_f16(
                        false, af[i], false, bfr[j],
                        (short)0, acc[i][j], false, false);
        }
        __syncthreads();

        xp += BK; wp += 64; sp += 1; zp += 1;
    }

    float bv[4];
    #pragma unroll
    for (int j = 0; j < 4; ++j)
        bv[j] = bias[n0 + wn + j * 16 + r];
    #pragma unroll
    for (int i = 0; i < 4; ++i)
        #pragma unroll
        for (int j = 0; j < 4; ++j) {
            float* o = out + (size_t)(m0 + wm + i * 16 + hh * 8) * NDIM
                           + (n0 + wn + j * 16 + r);
            #pragma unroll
            for (int v = 0; v < 8; ++v)
                __builtin_nontemporal_store(acc[i][j][v] + bv[j], o + (size_t)v * NDIM);
        }
}

// ============================================================================
extern "C" void kernel_launch(void* const* d_in, const int* in_sizes, int n_in,
                              void* d_out, int out_size, void* d_ws, size_t ws_size,
                              hipStream_t stream) {
    const float* x       = (const float*)d_in[0];
    const int*   Wq      = (const int*)d_in[1];
    const int*   scale_q = (const int*)d_in[2];
    const int*   zero_q  = (const int*)d_in[3];
    const float* s_scale = (const float*)d_in[4];
    const float* z_scale = (const float*)d_in[5];
    const float* s_zero  = (const float*)d_in[6];
    const float* z_zero  = (const float*)d_in[7];
    const float* bias    = (const float*)d_in[8];
    float* out = (float*)d_out;

    const size_t xh_bytes = (size_t)MDIM * KDIM * sizeof(_Float16);  // 64 MiB
    const size_t wh_bytes = (size_t)NDIM * KDIM * sizeof(_Float16);  // 32 MiB
    dim3 grid(NDIM / BN, MDIM / BM);   // 32 x 32

    if (ws_size >= xh_bytes + wh_bytes) {
        _Float16* xh = (_Float16*)d_ws;
        _Float16* wh = (_Float16*)((char*)d_ws + xh_bytes);
        cvt_x_kernel<<<(MDIM * KDIM) / (256 * 8), 256, 0, stream>>>(x, xh);
        deq_w_kernel<<<(NDIM * KDIM) / (256 * 16), 256, 0, stream>>>(
            Wq, scale_q, zero_q, s_scale, z_scale, s_zero, z_zero, wh);
        gemm_tdm_kernel<<<grid, dim3(256), 0, stream>>>(xh, wh, bias, out);
    } else {
        hqq_wmma_fused_kernel<<<grid, dim3(256), 0, stream>>>(
            x, Wq, scale_q, zero_q, s_scale, z_scale, s_zero, z_zero, bias, out);
    }
}
